// NeuroISNet_9672266350947
// MI455X (gfx1250) — matrix operand order repaired and checked by reading, not compiled, not run on hardware
//
#include <hip/hip_runtime.h>
#include <hip/hip_bf16.h>

typedef __attribute__((ext_vector_type(4)))  _Float16 v4h;
typedef __attribute__((ext_vector_type(8)))  _Float16 v8h;
typedef __attribute__((ext_vector_type(16))) _Float16 v16h;
typedef __attribute__((ext_vector_type(8)))  float    v8f;

#define B_  4
#define NN_ 4096
#define HD_ 128
#define ITERS_ 8
#define MTOT_ (B_*NN_)

// ---------------------------------------------------------------- helpers

__device__ __forceinline__ float sigm(float x) { return 1.0f / (1.0f + __expf(-x)); }

__device__ __forceinline__ float block_sum(float v, float* red) {
  int lane = threadIdx.x & 31, wid = threadIdx.x >> 5;
  int nw = blockDim.x >> 5;
  #pragma unroll
  for (int off = 16; off; off >>= 1) v += __shfl_xor(v, off, 32);
  if (lane == 0) red[wid] = v;
  __syncthreads();
  float s = 0.f;
  for (int i = 0; i < nw; ++i) s += red[i];
  __syncthreads();
  return s;
}

// CDNA5 async global->LDS staging (ASYNCcnt-tracked; no VGPR data path).
__device__ __forceinline__ void async_ld16(unsigned lds_off, const _Float16* g) {
  asm volatile("global_load_async_to_lds_b128 %0, %1, off"
               :: "v"(lds_off), "v"((unsigned long long)(size_t)g) : "memory");
}
__device__ __forceinline__ void wait_async4() { asm volatile("s_wait_asynccnt 0x4" ::: "memory"); }
__device__ __forceinline__ void wait_async0() { asm volatile("s_wait_asynccnt 0x0" ::: "memory"); }

// A fragment: 16x32 f16.
// lane l: row m = l&15 ; K = (l>>4)*16 + 0..15 (contiguous)
__device__ __forceinline__ v16h load_a_frag(const _Float16* __restrict__ A, int ld, int lane) {
  int m = lane & 15, half = lane >> 4;
  const _Float16* p = A + (size_t)m * ld + half * 16;
  v8h lo = *(const v8h*)p;
  v8h hi = *(const v8h*)(p + 8);
  v16h a;
  #pragma unroll
  for (int i = 0; i < 8; ++i) { a[i] = lo[i]; a[i + 8] = hi[i]; }
  return a;
}

// B fragment: 32x16 f16. lane l: col n = l&15 ; K = (l>>4)*16 + 0..15 (contiguous).
__device__ __forceinline__ v16h load_b_frag(const _Float16* __restrict__ Wt, size_t ld, int lane) {
  int n = lane & 15, kh = lane >> 4;
  const _Float16* p = Wt + (size_t)n * ld + kh * 16;
  v8h lo = *(const v8h*)p;
  v8h hi = *(const v8h*)(p + 8);
  v16h b;
  #pragma unroll
  for (int i = 0; i < 8; ++i) { b[i] = lo[i]; b[i + 8] = hi[i]; }
  return b;
}

// ---------------------------------------------------------------- f32 -> f16 convert

__global__ void cvt_kernel(const float* __restrict__ in, _Float16* __restrict__ out, size_t n4) {
  size_t stride = (size_t)gridDim.x * blockDim.x;
  for (size_t i = (size_t)blockIdx.x * blockDim.x + threadIdx.x; i < n4; i += stride) {
    float4 v = ((const float4*)in)[i];
    v4h o; o[0] = (_Float16)v.x; o[1] = (_Float16)v.y; o[2] = (_Float16)v.z; o[3] = (_Float16)v.w;
    ((v4h*)out)[i] = o;
  }
}

__global__ void gbias_kernel(const float* __restrict__ bih, const float* __restrict__ bhh,
                             float* __restrict__ gb) {
  int i = blockIdx.x * blockDim.x + threadIdx.x;
  if (i < 4 * HD_) gb[i] = bih[i] + bhh[i];
}

// ---------------------------------------------------------------- init MLP + LN (tiny)

__global__ void init_kernel(const float* __restrict__ kv, const float* __restrict__ nv,
                            const float* __restrict__ w1, const float* __restrict__ b1,
                            const float* __restrict__ w2, const float* __restrict__ b2,
                            const float* __restrict__ w3, const float* __restrict__ b3,
                            const float* __restrict__ lng, const float* __restrict__ lnb,
                            _Float16* __restrict__ h0h, _Float16* __restrict__ e0h) {
  __shared__ float sh[HD_];
  __shared__ float red[4];
  int b = blockIdx.x, t = threadIdx.x;
  float i0 = kv[b], i1 = nv[b];
  float a1 = fmaxf(w1[t * 2 + 0] * i0 + w1[t * 2 + 1] * i1 + b1[t], 0.f);
  sh[t] = a1; __syncthreads();
  float a2 = b2[t];
  for (int j = 0; j < HD_; ++j) a2 += w2[t * HD_ + j] * sh[j];
  a2 = fmaxf(a2, 0.f);
  __syncthreads(); sh[t] = a2; __syncthreads();
  float a3 = b3[t];
  for (int j = 0; j < HD_; ++j) a3 += w3[t * HD_ + j] * sh[j];
  h0h[b * HD_ + t] = (_Float16)a3;
  float mu = block_sum(a3, red) * (1.0f / HD_);
  float d = a3 - mu;
  float var = block_sum(d * d, red) * (1.0f / HD_);
  float y = d * rsqrtf(var + 1e-5f) * lng[t] + lnb[t];
  e0h[b * HD_ + t] = (_Float16)y;
}

__global__ void bcast_kernel(const _Float16* __restrict__ h0, const _Float16* __restrict__ e0,
                             _Float16* __restrict__ h, _Float16* __restrict__ e,
                             float* __restrict__ c, size_t total) {
  size_t stride = (size_t)gridDim.x * blockDim.x;
  for (size_t idx = (size_t)blockIdx.x * blockDim.x + threadIdx.x; idx < total; idx += stride) {
    int j = (int)(idx & (HD_ - 1));
    int b = (int)(idx / ((size_t)NN_ * HD_));
    h[idx] = h0[b * HD_ + j];
    e[idx] = e0[b * HD_ + j];
    c[idx] = 0.f;
  }
}

// ---------------------------------------------------------------- K=128 WMMA GEMM
// C[M x Nout] = act( A*W^T (+ A2*W2^T) + bias ); each wave: 16 rows x 64 cols.
// OMODE: 0 = f16 [M x Nout], 1 = f32 [M x Nout], 2 = f16 transposed [Nout x M]

template <bool HAS2, bool RELU, int OMODE>
__global__ __launch_bounds__(256) void gemm128_kernel(
    const _Float16* __restrict__ A, const _Float16* __restrict__ A2,
    const _Float16* __restrict__ W, const _Float16* __restrict__ W2,
    const float* __restrict__ bias, void* __restrict__ out,
    int M, int Nout) {
  const int K = HD_;
  int lane = threadIdx.x & 31;
  int wid = (blockIdx.x * blockDim.x + threadIdx.x) >> 5;
  int ngroups = Nout >> 6;
  int mt = wid / ngroups, ng = wid % ngroups;
  int row0 = mt * 16, col0 = ng * 64;
  if (row0 >= M) return;

  v8f acc[4] = {};
  #pragma unroll
  for (int kk = 0; kk < K; kk += 32) {
    v16h a = load_a_frag(A + (size_t)row0 * K + kk, K, lane);
    #pragma unroll
    for (int nt = 0; nt < 4; ++nt) {
      v16h bw = load_b_frag(W + (size_t)(col0 + nt * 16) * K + kk, K, lane);
      acc[nt] = __builtin_amdgcn_wmma_f32_16x16x32_f16(false, a, false, bw,
                                                       (short)0, acc[nt], false, false);
    }
    if constexpr (HAS2) {
      v16h a2 = load_a_frag(A2 + (size_t)row0 * K + kk, K, lane);
      #pragma unroll
      for (int nt = 0; nt < 4; ++nt) {
        v16h bw2 = load_b_frag(W2 + (size_t)(col0 + nt * 16) * K + kk, K, lane);
        acc[nt] = __builtin_amdgcn_wmma_f32_16x16x32_f16(false, a2, false, bw2,
                                                         (short)0, acc[nt], false, false);
      }
    }
  }
  int n_in = lane & 15, mh = lane >> 4;
  #pragma unroll
  for (int nt = 0; nt < 4; ++nt) {
    int col = col0 + nt * 16 + n_in;
    float bv = bias[col];
    #pragma unroll
    for (int v = 0; v < 8; ++v) {
      int row = row0 + v + mh * 8;
      float val = acc[nt][v] + bv;
      if constexpr (RELU) val = fmaxf(val, 0.f);
      if constexpr (OMODE == 0) ((_Float16*)out)[(size_t)row * Nout + col] = (_Float16)val;
      else if constexpr (OMODE == 1) ((float*)out)[(size_t)row * Nout + col] = val;
      else ((_Float16*)out)[(size_t)col * M + row] = (_Float16)val;
    }
  }
}

// ---------------------------------------------------------------- aggregation GEMM
// msg[b] = x[b] (NxN, f16) @ m[b] (NxH); B operand is mT[H][MTOT] (K-contiguous).
// Block: 128 rows x 128 cols. K tiled by 32 through double-buffered LDS filled with
// GLOBAL_LOAD_ASYNC_TO_LDS_B128. Loop unrolled x2 so the buffer index is a
// compile-time constant in each half (no accumulator/address phis).

#define LDT 40   // padded f16 stride (80 B): rows land in distinct bank groups

__global__ __launch_bounds__(256) void agg_kernel(
    const _Float16* __restrict__ Xh, const _Float16* __restrict__ mT,
    _Float16* __restrict__ msgH) {
  __shared__ _Float16 xs[2][128 * LDT];
  __shared__ _Float16 bs[2][128 * LDT];
  int b = blockIdx.y;
  int mbase = blockIdx.x * 128;
  int tid = threadIdx.x, wid = tid >> 5, lane = tid & 31;
  const _Float16* Xb = Xh + (size_t)b * NN_ * NN_;
  size_t colbase = (size_t)b * NN_;

  int r = tid >> 1;          // 0..127: staged row (x) / staged h-row (mT)
  int hh = tid & 1;          // which 16-f16 chunk of the 32-wide K tile
  int soff = r * LDT + hh * 16;

  v8f acc[8] = {};

  // issue 4 async b128 loads (32 B of x row + 32 B of mT row) into buffer `buf`
  auto issue_tile = [&](int kk, int buf) {
    const _Float16* gx = Xb + (size_t)(mbase + r) * NN_ + kk + hh * 16;
    const _Float16* gm = mT + (size_t)r * MTOT_ + colbase + kk + hh * 16;
    unsigned lx = (unsigned)(size_t)(const void*)&xs[buf][soff];
    unsigned lm = (unsigned)(size_t)(const void*)&bs[buf][soff];
    async_ld16(lx,      gx);
    async_ld16(lx + 16, gx + 8);
    async_ld16(lm,      gm);
    async_ld16(lm + 16, gm + 8);
  };

  // consume the tile resident in buffer `cur` (cur is a compile-time constant)
  auto compute_tile = [&](int cur) {
    v16h a;
    {
      int m = wid * 16 + (lane & 15), half = lane >> 4;
      const _Float16* p = &xs[cur][m * LDT + half * 16];
      v8h lo = *(const v8h*)p;
      v8h hi = *(const v8h*)(p + 8);
      #pragma unroll
      for (int i = 0; i < 8; ++i) { a[i] = lo[i]; a[i + 8] = hi[i]; }
    }
    #pragma unroll
    for (int nt = 0; nt < 8; ++nt) {
      int n = nt * 16 + (lane & 15), kh = lane >> 4;
      const _Float16* p = &bs[cur][n * LDT + kh * 16];
      v8h lo = *(const v8h*)p;
      v8h hi = *(const v8h*)(p + 8);
      v16h bw;
      #pragma unroll
      for (int i = 0; i < 8; ++i) { bw[i] = lo[i]; bw[i + 8] = hi[i]; }
      acc[nt] = __builtin_amdgcn_wmma_f32_16x16x32_f16(false, a, false, bw,
                                                       (short)0, acc[nt], false, false);
    }
  };

  issue_tile(0, 0);
  for (int kk = 0; kk < NN_; kk += 64) {   // two 32-wide tiles per trip (128 tiles total)
    // ---- tile in buffer 0 (load next into buffer 1)
    if (kk + 32 < NN_) { issue_tile(kk + 32, 1); wait_async4(); }
    else               { wait_async0(); }
    __syncthreads();
    compute_tile(0);
    __syncthreads();
    // ---- tile in buffer 1 (load next into buffer 0)
    if (kk + 64 < NN_) { issue_tile(kk + 64, 0); wait_async4(); }
    else               { wait_async0(); }
    __syncthreads();
    compute_tile(1);
    __syncthreads();
  }

  int n_in = lane & 15, mh = lane >> 4;
  #pragma unroll
  for (int nt = 0; nt < 8; ++nt)
    #pragma unroll
    for (int v = 0; v < 8; ++v) {
      int row = mbase + wid * 16 + v + mh * 8;
      msgH[((size_t)b * NN_ + row) * HD_ + nt * 16 + n_in] = (_Float16)acc[nt][v];
    }
}

// ---------------------------------------------------------------- fused LSTM cell + LayerNorm

__global__ void lstm_ln_kernel(const float* __restrict__ gates, float* __restrict__ c,
                               _Float16* __restrict__ hH, _Float16* __restrict__ eH,
                               const float* __restrict__ lng, const float* __restrict__ lnb) {
  __shared__ float red[4];
  int row = blockIdx.x, t = threadIdx.x;
  size_t gb = (size_t)row * (4 * HD_);
  float gi = gates[gb + t], gf = gates[gb + HD_ + t];
  float gg = gates[gb + 2 * HD_ + t], go = gates[gb + 3 * HD_ + t];
  size_t idx = (size_t)row * HD_ + t;
  float cn = sigm(gf) * c[idx] + sigm(gi) * tanhf(gg);
  c[idx] = cn;
  float hv = sigm(go) * tanhf(cn);
  hH[idx] = (_Float16)hv;
  float mu = block_sum(hv, red) * (1.0f / HD_);
  float d = hv - mu;
  float var = block_sum(d * d, red) * (1.0f / HD_);
  eH[idx] = (_Float16)(d * rsqrtf(var + 1e-5f) * lng[t] + lnb[t]);
}

// ---------------------------------------------------------------- vote tail

__global__ void zero_kernel(float* __restrict__ acc) {
  if (threadIdx.x < B_) acc[threadIdx.x] = 0.f;
}

__global__ void vote_kernel(const _Float16* __restrict__ v2, const float* __restrict__ w3,
                            const float* __restrict__ b3, const float* __restrict__ mask,
                            float* __restrict__ acc) {
  __shared__ float red[8];
  int blocksPerBatch = NN_ / 256;
  int b = blockIdx.x / blocksPerBatch;
  int rl = (blockIdx.x % blocksPerBatch) * 256 + threadIdx.x;
  int row = b * NN_ + rl;
  const _Float16* p = v2 + (size_t)row * HD_;
  float dot = b3[0];
  #pragma unroll 8
  for (int j = 0; j < HD_; ++j) dot += (float)p[j] * w3[j];
  float s = block_sum(dot * mask[row], red);
  if (threadIdx.x == 0) atomicAdd(&acc[b], s);
}

__global__ void final_kernel(const float* __restrict__ acc, float* __restrict__ out) {
  if (threadIdx.x < B_) out[threadIdx.x] = sigm(acc[threadIdx.x]);
}

// ---------------------------------------------------------------- host

extern "C" void kernel_launch(void* const* d_in, const int* in_sizes, int n_in,
                              void* d_out, int out_size, void* d_ws, size_t ws_size,
                              hipStream_t stream) {
  const float* x    = (const float*)d_in[0];
  const float* kv   = (const float*)d_in[1];
  const float* nv   = (const float*)d_in[2];
  const float* mask = (const float*)d_in[3];
  const float* iw1 = (const float*)d_in[4],  *ib1 = (const float*)d_in[5];
  const float* iw2 = (const float*)d_in[6],  *ib2 = (const float*)d_in[7];
  const float* iw3 = (const float*)d_in[8],  *ib3 = (const float*)d_in[9];
  const float* mw1 = (const float*)d_in[10], *mb1 = (const float*)d_in[11];
  const float* mw2 = (const float*)d_in[12], *mb2 = (const float*)d_in[13];
  const float* mw3 = (const float*)d_in[14], *mb3 = (const float*)d_in[15];
  const float* wih = (const float*)d_in[16], *whh = (const float*)d_in[17];
  const float* bih = (const float*)d_in[18], *bhh = (const float*)d_in[19];
  const float* lng = (const float*)d_in[20], *lnb = (const float*)d_in[21];
  const float* vw1 = (const float*)d_in[22], *vb1 = (const float*)d_in[23];
  const float* vw2 = (const float*)d_in[24], *vb2 = (const float*)d_in[25];
  const float* vw3 = (const float*)d_in[26], *vb3 = (const float*)d_in[27];
  float* out = (float*)d_out;

  char* ws = (char*)d_ws;
  size_t off = 0;
  auto alloc = [&](size_t bytes) { size_t o = off; off = (off + bytes + 255) & ~(size_t)255; return o; };
  _Float16* xh   = (_Float16*)(ws + alloc((size_t)B_ * NN_ * NN_ * 2));
  _Float16* wm1h = (_Float16*)(ws + alloc(HD_ * HD_ * 2));
  _Float16* wm2h = (_Float16*)(ws + alloc(HD_ * HD_ * 2));
  _Float16* wm3h = (_Float16*)(ws + alloc(HD_ * HD_ * 2));
  _Float16* wihh = (_Float16*)(ws + alloc(4 * HD_ * HD_ * 2));
  _Float16* whhh = (_Float16*)(ws + alloc(4 * HD_ * HD_ * 2));
  _Float16* wv1h = (_Float16*)(ws + alloc(HD_ * HD_ * 2));
  _Float16* wv2h = (_Float16*)(ws + alloc(HD_ * HD_ * 2));
  float*    gbias = (float*)(ws + alloc(4 * HD_ * 4));
  _Float16* h0h  = (_Float16*)(ws + alloc(B_ * HD_ * 2));
  _Float16* e0h  = (_Float16*)(ws + alloc(B_ * HD_ * 2));
  _Float16* hH   = (_Float16*)(ws + alloc((size_t)MTOT_ * HD_ * 2));
  _Float16* eH   = (_Float16*)(ws + alloc((size_t)MTOT_ * HD_ * 2));
  float*    cF   = (float*)(ws + alloc((size_t)MTOT_ * HD_ * 4));
  _Float16* t1H  = (_Float16*)(ws + alloc((size_t)MTOT_ * HD_ * 2));
  _Float16* t2H  = (_Float16*)(ws + alloc((size_t)MTOT_ * HD_ * 2));
  _Float16* mT   = (_Float16*)(ws + alloc((size_t)HD_ * MTOT_ * 2));
  _Float16* msgH = (_Float16*)(ws + alloc((size_t)MTOT_ * HD_ * 2));
  float*    gatesF = (float*)(ws + alloc((size_t)MTOT_ * 4 * HD_ * 4));
  float*    accB = (float*)(ws + alloc(B_ * 4));
  (void)ws_size; (void)n_in; (void)in_sizes; (void)out_size;

  // --- one-time prep: f16 copies of x and weights, combined LSTM bias
  cvt_kernel<<<8192, 256, 0, stream>>>(x, xh, (size_t)B_ * NN_ * NN_ / 4);
  cvt_kernel<<<16, 256, 0, stream>>>(mw1, wm1h, HD_ * HD_ / 4);
  cvt_kernel<<<16, 256, 0, stream>>>(mw2, wm2h, HD_ * HD_ / 4);
  cvt_kernel<<<16, 256, 0, stream>>>(mw3, wm3h, HD_ * HD_ / 4);
  cvt_kernel<<<64, 256, 0, stream>>>(wih, wihh, 4 * HD_ * HD_ / 4);
  cvt_kernel<<<64, 256, 0, stream>>>(whh, whhh, 4 * HD_ * HD_ / 4);
  cvt_kernel<<<16, 256, 0, stream>>>(vw1, wv1h, HD_ * HD_ / 4);
  cvt_kernel<<<16, 256, 0, stream>>>(vw2, wv2h, HD_ * HD_ / 4);
  gbias_kernel<<<2, 256, 0, stream>>>(bih, bhh, gbias);

  // --- init MLP + broadcast state
  init_kernel<<<B_, HD_, 0, stream>>>(kv, nv, iw1, ib1, iw2, ib2, iw3, ib3, lng, lnb, h0h, e0h);
  bcast_kernel<<<4096, 256, 0, stream>>>(h0h, e0h, hH, eH, cF, (size_t)MTOT_ * HD_);

  const int gemmH_blocks  = (MTOT_ / 16) * (HD_ / 64) / 8;       // 256
  const int gemm4H_blocks = (MTOT_ / 16) * (4 * HD_ / 64) / 8;   // 1024

  for (int it = 0; it < ITERS_; ++it) {
    // msg MLP: embed -> t1 -> t2 -> mT (transposed output for aggregation)
    gemm128_kernel<false, true, 0><<<gemmH_blocks, 256, 0, stream>>>(
        eH, nullptr, wm1h, nullptr, mb1, t1H, MTOT_, HD_);
    gemm128_kernel<false, true, 0><<<gemmH_blocks, 256, 0, stream>>>(
        t1H, nullptr, wm2h, nullptr, mb2, t2H, MTOT_, HD_);
    gemm128_kernel<false, false, 2><<<gemmH_blocks, 256, 0, stream>>>(
        t2H, nullptr, wm3h, nullptr, mb3, mT, MTOT_, HD_);
    // dense aggregation: msg = x @ m  (async-LDS double-buffered WMMA)
    agg_kernel<<<dim3(NN_ / 128, B_), 256, 0, stream>>>(xh, mT, msgH);
    // gates = msg@wih^T + h@whh^T + (bih+bhh)
    gemm128_kernel<true, false, 1><<<gemm4H_blocks, 256, 0, stream>>>(
        msgH, hH, wihh, whhh, gbias, gatesF, MTOT_, 4 * HD_);
    // LSTM cell update + LayerNorm -> h (f16), embed (f16), c (f32)
    lstm_ln_kernel<<<MTOT_, HD_, 0, stream>>>(gatesF, cF, hH, eH, lng, lnb);
  }

  // vote MLP + masked reduction + sigmoid
  gemm128_kernel<false, true, 0><<<gemmH_blocks, 256, 0, stream>>>(
      eH, nullptr, wv1h, nullptr, vb1, t1H, MTOT_, HD_);
  gemm128_kernel<false, true, 0><<<gemmH_blocks, 256, 0, stream>>>(
      t1H, nullptr, wv2h, nullptr, vb2, t2H, MTOT_, HD_);
  zero_kernel<<<1, 32, 0, stream>>>(accB);
  vote_kernel<<<B_ * (NN_ / 256), 256, 0, stream>>>(t2H, vw3, vb3, mask, accB);
  final_kernel<<<1, 32, 0, stream>>>(accB, out);
}
